// QSimNet_68247030333457
// MI455X (gfx1250) — matrix operand order repaired and checked by reading, not compiled, not run on hardware
//
#include <hip/hip_runtime.h>
#include <hip/hip_bf16.h>
#include <cstdint>
#include <cstddef>

// ---------------------------------------------------------------------------
// Problem constants (from the reference)
// ---------------------------------------------------------------------------
static constexpr int NN   = 131072;   // nodes
static constexpr int QN   = 4096;     // graphs / queries
static constexpr int NPG  = 32;       // nodes per graph
static constexpr int EPG  = 64;       // edges per graph
static constexpr int DD   = 256;      // hidden dim
static constexpr int MEMN = 2048;     // memory slots
static constexpr int MAXK = 16;

typedef __attribute__((ext_vector_type(16))) __bf16 v16bf;
typedef __attribute__((ext_vector_type(8)))  float  v8f;

// ---------------------------------------------------------------------------
// Async global->LDS (CDNA5 GLOBAL_LOAD_ASYNC_TO_LDS_B128, ASYNCcnt-tracked).
// Builtin signature (from hipcc diagnostic): first arg is
// 'int __vector_size__(16) __device__ *'  == v4i addrspace(1)*, second is the
// LDS-side v4i addrspace(3)*. Guarded so the file still compiles elsewhere.
// ---------------------------------------------------------------------------
#if defined(__HIP_DEVICE_COMPILE__) && \
    __has_builtin(__builtin_amdgcn_global_load_async_to_lds_b128) && \
    __has_builtin(__builtin_amdgcn_s_wait_asynccnt)
#define HAVE_ASYNC_LDS 1
#else
#define HAVE_ASYNC_LDS 0
#endif

#if HAVE_ASYNC_LDS
typedef int v4i_vs __attribute__((vector_size(16)));
typedef __attribute__((address_space(1))) v4i_vs* gv4i_t;
typedef __attribute__((address_space(3))) v4i_vs* lv4i_t;
__device__ __forceinline__ gv4i_t as_g128(const void* p) {
    return (gv4i_t)(uintptr_t)p;                 // AS1 pointer: full 64-bit address
}
__device__ __forceinline__ lv4i_t as_l128(const void* p) {
    return (lv4i_t)(uint32_t)(uintptr_t)p;       // AS3 pointer: low 32 bits = LDS offset
}
#endif

// ---------------------------------------------------------------------------
// bf16 WMMA GEMM:  C[M,Ncol] = act(A[M,K] * Bt[Ncol,K]^T + bias)
// Bt is the N-major (pre-transposed) operand so both tiles stage with b128.
// Block: 256 threads (8 waves). Tile: BM=128, BN=64, BK=32.
// Each wave computes a 32x32 patch = 2x2 v_wmma_f32_16x16x32_bf16 tiles.
// M % 128 == 0, Ncol % 64 == 0, K % 32 == 0 for every call in this pipeline.
// ---------------------------------------------------------------------------
#define BM 128
#define BN 64
#define BK 32
#define LDPAD 8   // pad to 40 bf16/row (80B): keeps 16B alignment, spreads banks

template <bool RELU, bool OUT_BF16>
__global__ __launch_bounds__(256) void gemm_bf16_kernel(
    const __bf16* __restrict__ A, const __bf16* __restrict__ Bt,
    const float* __restrict__ bias, void* __restrict__ Cout,
    int M, int Ncol, int K)
{
    __shared__ __bf16 As[BM][BK + LDPAD];   // 128*40*2 = 10240 B
    __shared__ __bf16 Bts[BN][BK + LDPAD];  //  64*40*2 =  5120 B

    const int tid  = threadIdx.x;
    const int lane = tid & 31;
    const int wave = tid >> 5;       // 0..7
    const int wm   = wave >> 1;      // 0..3 : 32-row group
    const int wn   = wave & 1;       // 0..1 : 32-col group
    const int lr   = lane & 15;
    const int lh   = lane >> 4;      // K-half select
    const long rowBase = (long)blockIdx.x * BM;
    const long colBase = (long)blockIdx.y * BN;

    // per-thread staging assignments
    const int ar = tid >> 1, ah = tid & 1;   // A: row 0..127, 16-elem half
    const int br = tid >> 2, bq = tid & 3;   // Bt: row 0..63, 8-elem quarter

    v8f acc[2][2] = {};
    union AV { uint4 u[2]; v16bf v; };

    for (int k0 = 0; k0 < K; k0 += BK) {
        const __bf16* aSrc = A  + (size_t)(rowBase + ar) * K + k0 + ah * 16;
        const __bf16* bSrc = Bt + (size_t)(colBase + br) * K + k0 + bq * 8;
        __bf16* aDst = &As[ar][ah * 16];
        __bf16* bDst = &Bts[br][bq * 8];
#if HAVE_ASYNC_LDS
        // global -> LDS without VGPR round-trip; tracked by ASYNCcnt
        __builtin_amdgcn_global_load_async_to_lds_b128(as_g128(aSrc),     as_l128(aDst),     0, 0);
        __builtin_amdgcn_global_load_async_to_lds_b128(as_g128(aSrc + 8), as_l128(aDst + 8), 0, 0);
        __builtin_amdgcn_global_load_async_to_lds_b128(as_g128(bSrc),     as_l128(bDst),     0, 0);
        if (k0 + BK < K)
            __builtin_prefetch(A + (size_t)(rowBase + ar) * K + k0 + BK, 0, 0);
        __builtin_amdgcn_s_wait_asynccnt(0);
#else
        {
            const uint4* s0 = (const uint4*)aSrc;
            uint4* d0 = (uint4*)aDst;
            d0[0] = s0[0];
            d0[1] = s0[1];
            *(uint4*)bDst = *(const uint4*)bSrc;
            if (k0 + BK < K)
                __builtin_prefetch(A + (size_t)(rowBase + ar) * K + k0 + BK, 0, 0);
        }
#endif
        __syncthreads();

        // --- fragments (per ISA 16-bit A / B VGPR layouts, wave32)
        AV af[2], bfr[2];
#pragma unroll
        for (int i = 0; i < 2; ++i) {
            const __bf16* arp = &As[wm * 32 + i * 16 + lr][0];
            af[i].u[0] = *(const uint4*)(arp + lh * 8);        // K: lh*8 .. +7
            af[i].u[1] = *(const uint4*)(arp + 16 + lh * 8);   // K: 16+lh*8 .. +7
            const __bf16* brp = &Bts[wn * 32 + i * 16 + lr][0];
            bfr[i].u[0] = *(const uint4*)(brp + lh * 8);       // K: lh*8 .. +7
            bfr[i].u[1] = *(const uint4*)(brp + 16 + lh * 8);  // K: 16+lh*8 .. +7
        }
#pragma unroll
        for (int i = 0; i < 2; ++i)
#pragma unroll
            for (int j = 0; j < 2; ++j)
                acc[i][j] = __builtin_amdgcn_wmma_f32_16x16x32_bf16(
                    false, af[i].v, false, bfr[j].v, (short)0, acc[i][j], false, false);
        __syncthreads();
    }

    // --- epilogue: C layout -> lane holds col = lr, rows v + 8*lh per VGPR v
#pragma unroll
    for (int i = 0; i < 2; ++i) {
#pragma unroll
        for (int j = 0; j < 2; ++j) {
            const long col = colBase + wn * 32 + j * 16 + lr;
            const float bv = bias ? bias[col] : 0.0f;
#pragma unroll
            for (int v = 0; v < 8; ++v) {
                const long row = rowBase + wm * 32 + i * 16 + v + 8 * lh;
                float val = acc[i][j][v] + bv;
                if (RELU) val = fmaxf(val, 0.0f);
                if (OUT_BF16)
                    ((__bf16*)Cout)[(size_t)row * Ncol + col] = (__bf16)val;
                else
                    ((float*)Cout)[(size_t)row * Ncol + col] = val;
            }
        }
    }
}

// ---------------------------------------------------------------------------
// Elementwise / support kernels
// ---------------------------------------------------------------------------
__global__ void cvt_f32_bf16_kernel(const float* __restrict__ s,
                                    __bf16* __restrict__ d, size_t n)
{
    size_t i = (size_t)blockIdx.x * 256 + threadIdx.x;
    if (i < n) d[i] = (__bf16)s[i];
}

// dst[c*R + r] = src[r*C + c]   (src is [R,C] row-major f32 -> dst is [C,R] bf16)
__global__ void transpose_cvt_kernel(const float* __restrict__ s,
                                     __bf16* __restrict__ d, int R, int C)
{
    long i = (long)blockIdx.x * 256 + threadIdx.x;
    if (i < (long)R * C) {
        int r = (int)(i / C), c = (int)(i % C);
        d[(size_t)c * R + r] = (__bf16)s[i];
    }
}

// WcT[n*D + k] = sum_i softmax(sim_w)[i] * sims[i].W[k*D + n]  (transposed out)
__global__ void combine_sims_kernel(const float* __restrict__ w4,
    const float* __restrict__ W0, const float* __restrict__ W1,
    const float* __restrict__ W2, const float* __restrict__ W3,
    const float* __restrict__ b0, const float* __restrict__ b1,
    const float* __restrict__ b2, const float* __restrict__ b3,
    __bf16* __restrict__ WcT, float* __restrict__ bc)
{
    long i = (long)blockIdx.x * 256 + threadIdx.x;
    float m = fmaxf(fmaxf(w4[0], w4[1]), fmaxf(w4[2], w4[3]));
    float e0 = expf(w4[0] - m), e1 = expf(w4[1] - m);
    float e2 = expf(w4[2] - m), e3 = expf(w4[3] - m);
    float s = e0 + e1 + e2 + e3;
    e0 /= s; e1 /= s; e2 /= s; e3 /= s;
    if (i < (long)DD * DD) {
        int k = (int)(i / DD), n = (int)(i % DD);
        WcT[(size_t)n * DD + k] =
            (__bf16)(e0 * W0[i] + e1 * W1[i] + e2 * W2[i] + e3 * W3[i]);
    }
    if (i < DD)
        bc[i] = e0 * b0[i] + e1 * b1[i] + e2 * b2[i] + e3 * b3[i];
}

// Per-graph GIN message + aggregation (block per graph, 256 threads).
// out[node] = bf16( h[node] + sum_{e: dst=node} relu(h[src] + ea@We + be) )
__global__ __launch_bounds__(256) void gin_msg_kernel(
    const float* __restrict__ h, const float* __restrict__ We,
    const float* __restrict__ be, const float* __restrict__ ea,
    const int* __restrict__ esrc, const int* __restrict__ edst,
    __bf16* __restrict__ out, int ind)
{
    __shared__ float agg[NPG * DD];     // 32 KB worst case
    __shared__ float We_s[16 * DD];     // 16 KB worst case
    __shared__ float ea_s[EPG * 16];    //  4 KB
    const int g = blockIdx.x, tid = threadIdx.x;

    for (int i = tid; i < NPG * ind; i += 256) agg[i] = 0.0f;
    for (int i = tid; i < 16 * ind;  i += 256) We_s[i] = We[i];
    for (int i = tid; i < EPG * 16;  i += 256) ea_s[i] = ea[(size_t)g * EPG * 16 + i];
    __syncthreads();

    const int c = tid;
    for (int j = 0; j < EPG; ++j) {
        const int e   = g * EPG + j;
        const int src = esrc[e];
        const int dl  = (edst[e] - g * NPG) & (NPG - 1);
        if (c < ind) {
            float m = h[(size_t)src * ind + c] + be[c];
#pragma unroll
            for (int f = 0; f < 16; ++f) m += ea_s[j * 16 + f] * We_s[f * ind + c];
            m = fmaxf(m, 0.0f);
            agg[dl * ind + c] += m;   // serial over edges, per-thread column: no race
        }
    }
    __syncthreads();
    if (c < ind) {
        for (int i = 0; i < NPG; ++i) {
            const size_t node = (size_t)g * NPG + i;
            out[node * ind + c] = (__bf16)(h[node * ind + c] + agg[i * ind + c]);
        }
    }
}

// Pool 32 nodes -> cat2[q] = bf16([s/32, s])  (block per graph, 256 threads)
__global__ __launch_bounds__(256) void pool_kernel(const float* __restrict__ h,
                                                   __bf16* __restrict__ cat2)
{
    const int q = blockIdx.x, c = threadIdx.x;
    float s = 0.0f;
    for (int i = 0; i < NPG; ++i) s += h[((size_t)q * NPG + i) * DD + c];
    cat2[(size_t)q * 2 * DD + c]      = (__bf16)(s / (float)NPG);
    cat2[(size_t)q * 2 * DD + DD + c] = (__bf16)s;
}

// wn = wemb / ||wemb|| -> bf16 row-major (serves as both A and Bt for S)
__global__ __launch_bounds__(256) void normalize_kernel(
    const float* __restrict__ wemb, __bf16* __restrict__ wn)
{
    __shared__ float red[256];
    const int q = blockIdx.x, c = threadIdx.x;
    const float v = wemb[(size_t)q * DD + c];
    red[c] = v * v;
    __syncthreads();
    for (int s2 = 128; s2 > 0; s2 >>= 1) {
        if (c < s2) red[c] += red[c + s2];
        __syncthreads();
    }
    wn[(size_t)q * DD + c] = (__bf16)(v * rsqrtf(red[0]));
}

// top-17 of S row (diag forced to 2.0) -> nbr = indices 1..16
__global__ __launch_bounds__(256) void topk_kernel(const float* __restrict__ S,
                                                   int* __restrict__ nbr)
{
    __shared__ float row[QN];   // 16 KB
    __shared__ float rv[256];
    __shared__ int   ri[256];
    __shared__ int   sel[MAXK + 1];
    const int q = blockIdx.x, tid = threadIdx.x;
    for (int c = tid; c < QN; c += 256)
        row[c] = (c == q) ? 2.0f : S[(size_t)q * QN + c];
    __syncthreads();
    for (int t = 0; t < MAXK + 1; ++t) {
        float bv = -3.0e38f; int bi = QN;
        for (int c = tid; c < QN; c += 256) {
            float v = row[c];
            if (v > bv || (v == bv && c < bi)) { bv = v; bi = c; }
        }
        rv[tid] = bv; ri[tid] = bi;
        __syncthreads();
        for (int s2 = 128; s2 > 0; s2 >>= 1) {
            if (tid < s2) {
                if (rv[tid + s2] > rv[tid] ||
                    (rv[tid + s2] == rv[tid] && ri[tid + s2] < ri[tid])) {
                    rv[tid] = rv[tid + s2]; ri[tid] = ri[tid + s2];
                }
            }
            __syncthreads();
        }
        if (tid == 0) { sel[t] = ri[0]; row[ri[0]] = -3.0e38f; }
        __syncthreads();
    }
    if (tid >= 1 && tid < MAXK + 1) nbr[(size_t)q * MAXK + (tid - 1)] = sel[tid];
}

// dyn_k = clip(int(sigmoid(cnt2(relu_hidden)) * 16), 1, 16)
__global__ void dynk_kernel(const __bf16* __restrict__ c1,
                            const float* __restrict__ W,
                            const float* __restrict__ b, int* __restrict__ dynk)
{
    const int q = blockIdx.x * 256 + threadIdx.x;
    if (q >= QN) return;
    float s = b[0];
    for (int j = 0; j < 128; ++j) s += (float)c1[(size_t)q * 128 + j] * W[j];
    const float sig = 1.0f / (1.0f + expf(-s));
    int k = (int)(sig * (float)MAXK);
    dynk[q] = k < 1 ? 1 : (k > MAXK ? MAXK : k);
}

// attention: scores, masked softmax, weighted neighbor aggregation
__global__ __launch_bounds__(256) void att_kernel(
    const float* __restrict__ P, const float* __restrict__ Pn,
    const float* __restrict__ att2W, const float* __restrict__ att2b,
    const int* __restrict__ nbr, const int* __restrict__ dynk,
    const float* __restrict__ emb, float* __restrict__ aggn)
{
    __shared__ float red[256];
    __shared__ float sc[MAXK];
    __shared__ int   nb[MAXK];
    const int q = blockIdx.x, c = threadIdx.x;
    if (c < MAXK) nb[c] = nbr[(size_t)q * MAXK + c];
    __syncthreads();
    const float pv = P[(size_t)q * DD + c];
    for (int k = 0; k < MAXK; ++k) {
        const float hcol = fmaxf(pv + Pn[(size_t)nb[k] * DD + c], 0.0f);
        red[c] = hcol * att2W[c];
        __syncthreads();
        for (int s2 = 128; s2 > 0; s2 >>= 1) {
            if (c < s2) red[c] += red[c + s2];
            __syncthreads();
        }
        if (c == 0) sc[k] = red[0] + att2b[0];
        __syncthreads();
    }
    if (c == 0) {
        const int kk = dynk[q];
        float mx = -3.0e38f;
        for (int k = 0; k < kk; ++k) mx = fmaxf(mx, sc[k]);
        float sm = 0.0f;
        for (int k = 0; k < MAXK; ++k) {
            float e = (k < kk) ? expf(sc[k] - mx) : 0.0f;
            sc[k] = e; sm += e;
        }
        for (int k = 0; k < MAXK; ++k) sc[k] /= sm;
    }
    __syncthreads();
    float a = 0.0f;
    for (int k = 0; k < MAXK; ++k) a += sc[k] * emb[(size_t)nb[k] * DD + c];
    aggn[(size_t)q * DD + c] = a;
}

// row softmax of (L * 10.0) over 2048 cols -> bf16 (memory attention weights)
__global__ __launch_bounds__(256) void memsoft_kernel(const float* __restrict__ L,
                                                      __bf16* __restrict__ mw)
{
    __shared__ float red[256];
    const int q = blockIdx.x, tid = threadIdx.x;
    const float* row = L + (size_t)q * MEMN;
    float mx = -3.0e38f;
    for (int c = tid; c < MEMN; c += 256) mx = fmaxf(mx, row[c] * 10.0f);
    red[tid] = mx;
    __syncthreads();
    for (int s2 = 128; s2 > 0; s2 >>= 1) {
        if (tid < s2) red[tid] = fmaxf(red[tid], red[tid + s2]);
        __syncthreads();
    }
    mx = red[0];
    __syncthreads();
    float sm = 0.0f;
    for (int c = tid; c < MEMN; c += 256) sm += expf(row[c] * 10.0f - mx);
    red[tid] = sm;
    __syncthreads();
    for (int s2 = 128; s2 > 0; s2 >>= 1) {
        if (tid < s2) red[tid] += red[tid + s2];
        __syncthreads();
    }
    const float inv = 1.0f / red[0];
    for (int c = tid; c < MEMN; c += 256)
        mw[(size_t)q * MEMN + c] = (__bf16)(expf(row[c] * 10.0f - mx) * inv);
}

__global__ void cat2x_kernel(const float* __restrict__ a, const float* __restrict__ b,
                             __bf16* __restrict__ out)
{
    const int q = blockIdx.x, c = threadIdx.x;
    out[(size_t)q * 2 * DD + c]      = (__bf16)a[(size_t)q * DD + c];
    out[(size_t)q * 2 * DD + DD + c] = (__bf16)b[(size_t)q * DD + c];
}

__global__ void cat3x_kernel(const float* __restrict__ a, const float* __restrict__ b,
                             const float* __restrict__ cc, __bf16* __restrict__ out)
{
    const int q = blockIdx.x, c = threadIdx.x;
    out[(size_t)q * 3 * DD + c]          = (__bf16)a[(size_t)q * DD + c];
    out[(size_t)q * 3 * DD + DD + c]     = (__bf16)b[(size_t)q * DD + c];
    out[(size_t)q * 3 * DD + 2 * DD + c] = (__bf16)cc[(size_t)q * DD + c];
}

__global__ void predfin_kernel(const __bf16* __restrict__ p2,
                               const float* __restrict__ W,
                               const float* __restrict__ b, float* __restrict__ out)
{
    const int q = blockIdx.x * 256 + threadIdx.x;
    if (q >= QN) return;
    float s = b[0];
    for (int j = 0; j < DD; ++j) s += (float)p2[(size_t)q * DD + j] * W[j];
    out[q] = s;
}

// ---------------------------------------------------------------------------
// Host-side launch helpers
// ---------------------------------------------------------------------------
static void launch_gemm(const __bf16* A, const __bf16* Bt, const float* bias,
                        void* C, int M, int Ncol, int K, bool relu, bool outBf16,
                        hipStream_t s)
{
    dim3 g((unsigned)(M / BM), (unsigned)(Ncol / BN)), b(256);
    if (relu) {
        if (outBf16) gemm_bf16_kernel<true,  true ><<<g, b, 0, s>>>(A, Bt, bias, C, M, Ncol, K);
        else         gemm_bf16_kernel<true,  false><<<g, b, 0, s>>>(A, Bt, bias, C, M, Ncol, K);
    } else {
        if (outBf16) gemm_bf16_kernel<false, true ><<<g, b, 0, s>>>(A, Bt, bias, C, M, Ncol, K);
        else         gemm_bf16_kernel<false, false><<<g, b, 0, s>>>(A, Bt, bias, C, M, Ncol, K);
    }
}

static void launch_cvt(const float* s, __bf16* d, size_t n, hipStream_t st)
{
    cvt_f32_bf16_kernel<<<(unsigned)((n + 255) / 256), 256, 0, st>>>(s, d, n);
}

// transpose-convert a [R,C] f32 weight into a [C,R] bf16 (the GEMM's Bt form)
static void launch_tcvt(const float* s, __bf16* d, int R, int C, hipStream_t st)
{
    transpose_cvt_kernel<<<(unsigned)(((size_t)R * C + 255) / 256), 256, 0, st>>>(s, d, R, C);
}

// ---------------------------------------------------------------------------
// kernel_launch
// Input order: jax pytree flattening of setup_inputs() (dicts sorted by key):
//  0 edge_attr, 1 edge_dst, 2 edge_src, 3 node2graph,
//  4/5 att1.W/b, 6/7 att2.W/b, 8/9 cnt1.W/b, 10/11 cnt2.W/b,
//  12..17 gin0{We,be,l1.W,l1.b,l2.W,l2.b}, 18..23 gin1{...}, 24..29 gin2{...},
//  30 mem_card (unused), 31 mem_emb, 32..37 pred1/2/3 {W,b},
//  38..41 proj1/proj2 {W,b}, 42 sim_w, 43..50 sims[0..3]{W,b},
//  51..54 upd1/upd2 {W,b}, 55 x
// ---------------------------------------------------------------------------
extern "C" void kernel_launch(void* const* d_in, const int* in_sizes, int n_in,
                              void* d_out, int out_size, void* d_ws, size_t ws_size,
                              hipStream_t stream)
{
    (void)in_sizes; (void)n_in; (void)out_size; (void)ws_size;

    const float* edge_attr = (const float*)d_in[0];
    const int*   edge_dst  = (const int*)d_in[1];
    const int*   edge_src  = (const int*)d_in[2];
    const float* att1W = (const float*)d_in[4];  const float* att1b = (const float*)d_in[5];
    const float* att2W = (const float*)d_in[6];  const float* att2b = (const float*)d_in[7];
    const float* cnt1W = (const float*)d_in[8];  const float* cnt1b = (const float*)d_in[9];
    const float* cnt2W = (const float*)d_in[10]; const float* cnt2b = (const float*)d_in[11];
    const float* ginWe[3]  = {(const float*)d_in[12], (const float*)d_in[18], (const float*)d_in[24]};
    const float* ginbe[3]  = {(const float*)d_in[13], (const float*)d_in[19], (const float*)d_in[25]};
    const float* ginl1W[3] = {(const float*)d_in[14], (const float*)d_in[20], (const float*)d_in[26]};
    const float* ginl1b[3] = {(const float*)d_in[15], (const float*)d_in[21], (const float*)d_in[27]};
    const float* ginl2W[3] = {(const float*)d_in[16], (const float*)d_in[22], (const float*)d_in[28]};
    const float* ginl2b[3] = {(const float*)d_in[17], (const float*)d_in[23], (const float*)d_in[29]};
    const float* memW   = (const float*)d_in[31];
    const float* pred1W = (const float*)d_in[32]; const float* pred1b = (const float*)d_in[33];
    const float* pred2W = (const float*)d_in[34]; const float* pred2b = (const float*)d_in[35];
    const float* pred3W = (const float*)d_in[36]; const float* pred3b = (const float*)d_in[37];
    const float* proj1W = (const float*)d_in[38]; const float* proj1b = (const float*)d_in[39];
    const float* proj2W = (const float*)d_in[40]; const float* proj2b = (const float*)d_in[41];
    const float* simw   = (const float*)d_in[42];
    const float* simW[4] = {(const float*)d_in[43], (const float*)d_in[45],
                            (const float*)d_in[47], (const float*)d_in[49]};
    const float* simb[4] = {(const float*)d_in[44], (const float*)d_in[46],
                            (const float*)d_in[48], (const float*)d_in[50]};
    const float* upd1W = (const float*)d_in[51]; const float* upd1b = (const float*)d_in[52];
    const float* upd2W = (const float*)d_in[53]; const float* upd2b = (const float*)d_in[54];
    const float* x     = (const float*)d_in[55];
    float* outp = (float*)d_out;

    // ---- workspace bump allocator (deterministic layout every call) ----
    char* base = (char*)d_ws;
    size_t off = 0;
    auto alloc = [&](size_t bytes) -> void* {
        void* r = base + off;
        off += (bytes + 255) & ~(size_t)255;
        return r;
    };

    // bf16 transposed weight copies ([out_dim, in_dim] = Bt form)
    __bf16* l1bf[3]; __bf16* l2bf[3];
    l1bf[0] = (__bf16*)alloc((size_t)64 * DD * 2);
    l1bf[1] = (__bf16*)alloc((size_t)DD * DD * 2);
    l1bf[2] = (__bf16*)alloc((size_t)DD * DD * 2);
    for (int l = 0; l < 3; ++l) l2bf[l] = (__bf16*)alloc((size_t)DD * DD * 2);
    __bf16* proj1bf = (__bf16*)alloc((size_t)2 * DD * DD * 2);   // [256,512]
    __bf16* proj2bf = (__bf16*)alloc((size_t)DD * DD * 2);
    __bf16* WcombbfT= (__bf16*)alloc((size_t)DD * DD * 2);
    float*  bcomb   = (float*) alloc((size_t)DD * 4);
    __bf16* cnt1bf  = (__bf16*)alloc((size_t)DD * 128 * 2);      // [128,256]
    __bf16* att1bf  = (__bf16*)alloc((size_t)2 * DD * DD * 2);   // topT then botT
    __bf16* upd1bf  = (__bf16*)alloc((size_t)2 * DD * DD * 2);   // [256,512]
    __bf16* upd2bf  = (__bf16*)alloc((size_t)DD * DD * 2);
    __bf16* pred1bf = (__bf16*)alloc((size_t)3 * DD * 2 * DD * 2); // [512,768]
    __bf16* pred2bf = (__bf16*)alloc((size_t)2 * DD * DD * 2);     // [256,512]
    __bf16* membf   = (__bf16*)alloc((size_t)MEMN * DD * 2);       // [2048,256]
    __bf16* memTbf  = (__bf16*)alloc((size_t)DD * MEMN * 2);       // [256,2048]

    // big activations
    float*  hbuf = (float*) alloc((size_t)NN * DD * 4);   // 134 MB
    __bf16* bufA = (__bf16*)alloc((size_t)NN * DD * 2);   //  67 MB (GIN in)
    __bf16* bufB = (__bf16*)alloc((size_t)NN * DD * 2);   //  67 MB (GIN mid)
    // lifetime aliasing: after GIN+pool, bufA/bufB are dead
    float*  Sbuf = (float*)(void*)bufA;                       // Q*Q f32 = 67 MB
    float*  Lbuf = (float*)(void*)bufB;                       // Q*2048 f32 = 33 MB
    __bf16* mwbf = (__bf16*)((char*)(void*)bufB + (size_t)QN * MEMN * 4);

    // small Q-sized buffers
    __bf16* cat2bf = (__bf16*)alloc((size_t)QN * 2 * DD * 2);
    __bf16* t1bf   = (__bf16*)alloc((size_t)QN * DD * 2);
    float*  emb    = (float*) alloc((size_t)QN * DD * 4);
    __bf16* embbf  = (__bf16*)alloc((size_t)QN * DD * 2);
    float*  wemb   = (float*) alloc((size_t)QN * DD * 4);
    __bf16* wnbf   = (__bf16*)alloc((size_t)QN * DD * 2);
    __bf16* c1bf   = (__bf16*)alloc((size_t)QN * 128 * 2);
    int*    dynk   = (int*)   alloc((size_t)QN * 4);
    int*    nbr    = (int*)   alloc((size_t)QN * MAXK * 4);
    float*  Pbuf   = (float*) alloc((size_t)QN * DD * 4);
    float*  Pnbuf  = (float*) alloc((size_t)QN * DD * 4);
    float*  aggn   = (float*) alloc((size_t)QN * DD * 4);
    float*  retr   = (float*) alloc((size_t)QN * DD * 4);
    __bf16* catUbf = (__bf16*)alloc((size_t)QN * 2 * DD * 2);
    __bf16* u1bf   = (__bf16*)alloc((size_t)QN * DD * 2);
    float*  updated= (float*) alloc((size_t)QN * DD * 4);
    __bf16* catPbf = (__bf16*)alloc((size_t)QN * 3 * DD * 2);
    __bf16* p1bf   = (__bf16*)alloc((size_t)QN * 2 * DD * 2);
    __bf16* p2bf   = (__bf16*)alloc((size_t)QN * DD * 2);

    // ---- 0) weight conversions (all transposed into Bt form) ----
    launch_tcvt(ginl1W[0], l1bf[0], 64, DD, stream);
    launch_tcvt(ginl1W[1], l1bf[1], DD, DD, stream);
    launch_tcvt(ginl1W[2], l1bf[2], DD, DD, stream);
    for (int l = 0; l < 3; ++l) launch_tcvt(ginl2W[l], l2bf[l], DD, DD, stream);
    launch_tcvt(proj1W, proj1bf, 2 * DD, DD, stream);
    launch_tcvt(proj2W, proj2bf, DD, DD, stream);
    launch_tcvt(cnt1W,  cnt1bf,  DD, 128, stream);
    launch_tcvt(att1W,                    att1bf,                    DD, DD, stream); // topT
    launch_tcvt(att1W + (size_t)DD * DD,  att1bf + (size_t)DD * DD,  DD, DD, stream); // botT
    launch_tcvt(upd1W,  upd1bf,  2 * DD, DD, stream);
    launch_tcvt(upd2W,  upd2bf,  DD, DD, stream);
    launch_tcvt(pred1W, pred1bf, 3 * DD, 2 * DD, stream);
    launch_tcvt(pred2W, pred2bf, 2 * DD, DD, stream);
    launch_cvt (memW,   membf,   (size_t)MEMN * DD, stream);       // Bt for logits
    launch_tcvt(memW,   memTbf,  MEMN, DD, stream);                // Bt for retrieval
    combine_sims_kernel<<<(unsigned)((DD * DD + 255) / 256), 256, 0, stream>>>(
        simw, simW[0], simW[1], simW[2], simW[3],
        simb[0], simb[1], simb[2], simb[3], WcombbfT, bcomb);

    // ---- 1) three GIN layers ----
    const float* hprev = x;
    for (int l = 0; l < 3; ++l) {
        const int ind = (l == 0) ? 64 : DD;
        gin_msg_kernel<<<QN, 256, 0, stream>>>(hprev, ginWe[l], ginbe[l], edge_attr,
                                               edge_src, edge_dst, bufA, ind);
        launch_gemm(bufA, l1bf[l], ginl1b[l], bufB, NN, DD, ind,  true, true,  stream);
        launch_gemm(bufB, l2bf[l], ginl2b[l], hbuf, NN, DD, DD,   true, false, stream);
        hprev = hbuf;
    }

    // ---- 2) pooling + projection -> emb ----
    pool_kernel<<<QN, 256, 0, stream>>>(hbuf, cat2bf);
    launch_gemm(cat2bf, proj1bf, proj1b, t1bf, QN, DD, 2 * DD, true,  true,  stream);
    launch_gemm(t1bf,   proj2bf, proj2b, emb,  QN, DD, DD,     false, false, stream);
    launch_cvt(emb, embbf, (size_t)QN * DD, stream);

    // ---- 3) similarity matrix S = wn @ wn^T (Bt of wn^T is wn itself) ----
    launch_gemm(embbf, WcombbfT, bcomb, wemb, QN, DD, DD, false, false, stream);
    normalize_kernel<<<QN, 256, 0, stream>>>(wemb, wnbf);
    launch_gemm(wnbf, wnbf, nullptr, Sbuf, QN, QN, DD, false, false, stream);

    // ---- 4) dynamic-k + top-k neighbors ----
    launch_gemm(embbf, cnt1bf, cnt1b, c1bf, QN, 128, DD, true, true, stream);
    dynk_kernel<<<(QN + 255) / 256, 256, 0, stream>>>(c1bf, cnt2W, cnt2b, dynk);
    topk_kernel<<<QN, 256, 0, stream>>>(Sbuf, nbr);

    // ---- 5) neighbor attention (att1 split into halves) ----
    launch_gemm(embbf, att1bf,                   att1b,   Pbuf,  QN, DD, DD, false, false, stream);
    launch_gemm(embbf, att1bf + (size_t)DD * DD, nullptr, Pnbuf, QN, DD, DD, false, false, stream);
    att_kernel<<<QN, 256, 0, stream>>>(Pbuf, Pnbuf, att2W, att2b, nbr, dynk, emb, aggn);

    // ---- 6) memory retrieval ----
    launch_gemm(embbf, membf, nullptr, Lbuf, QN, MEMN, DD, false, false, stream);
    memsoft_kernel<<<QN, 256, 0, stream>>>(Lbuf, mwbf);
    launch_gemm(mwbf, memTbf, nullptr, retr, QN, DD, MEMN, false, false, stream);

    // ---- 7) update MLP ----
    cat2x_kernel<<<QN, 256, 0, stream>>>(emb, retr, catUbf);
    launch_gemm(catUbf, upd1bf, upd1b, u1bf,    QN, DD, 2 * DD, true,  true,  stream);
    launch_gemm(u1bf,   upd2bf, upd2b, updated, QN, DD, DD,     false, false, stream);

    // ---- 8) prediction head ----
    cat3x_kernel<<<QN, 256, 0, stream>>>(emb, aggn, updated, catPbf);
    launch_gemm(catPbf, pred1bf, pred1b, p1bf, QN, 2 * DD, 3 * DD, true, true, stream);
    launch_gemm(p1bf,   pred2bf, pred2b, p2bf, QN, DD,     2 * DD, true, true, stream);
    predfin_kernel<<<(QN + 255) / 256, 256, 0, stream>>>(p2bf, pred3W, pred3b, outp);
}
